// Attention_42554535969268
// MI455X (gfx1250) — compile-verified, matrix operand order checked by reading, not run on hardware
//
#include <hip/hip_runtime.h>
#include <hip/hip_bf16.h>

// ---------------------------------------------------------------------------
// entmax-1.33 attention for MI455X (gfx1250), bf16 WMMA + LDS-resident bisection
// B=4, S=2048, HID=512, NH=8, DH=64, alpha=1.33, 50 bisection iterations
// ---------------------------------------------------------------------------

typedef __attribute__((ext_vector_type(16))) __bf16 v16bf;
typedef __attribute__((ext_vector_type(8)))  float  v8f;

#define BB        4
#define SS        2048
#define HIDN      512
#define NHEAD     8
#define DHEAD     64
#define ALPHA_M1  0.33f
#define PINV      (1.0f / ALPHA_M1)          // 1/(alpha-1) ~ 3.0303
#define QSCALE    (0.125f * ALPHA_M1)        // fold 1/sqrt(64) and (alpha-1) into Q
#define NITER     50
#define XA_STRIDE 2060                        // fp32 row stride (bank-conflict-free bisection)
#define PB_STRIDE 2056                        // bf16 row stride (16B aligned)

// ---- bf16 helpers (manual RNE convert; storage is unsigned short) ----------
static __device__ __forceinline__ unsigned short f2bf(float f) {
  unsigned int u = __float_as_uint(f);
  unsigned int r = u + 0x7FFFu + ((u >> 16) & 1u);
  return (unsigned short)(r >> 16);
}

static __device__ __forceinline__ unsigned int f2bf_pack2(float lo, float hi) {
  return (unsigned int)f2bf(lo) | ((unsigned int)f2bf(hi) << 16);
}

union FragU { uint4 u[2]; v16bf v; };

// 16-bit WMMA fragment: lane loads 8 bf16 at +0 and 8 bf16 at +16 elements.
// Caller pre-offsets the pointer by ((lane>>4)<<3) for the K-half split.
static __device__ __forceinline__ v16bf load_frag(const unsigned short* p) {
  FragU f;
  f.u[0] = *(const uint4*)(p);
  f.u[1] = *(const uint4*)(p + 16);
  return f.v;
}

static __device__ __forceinline__ v8f wmma_bf16(v16bf a, v16bf b, v8f c) {
  return __builtin_amdgcn_wmma_f32_16x16x32_bf16(false, a, false, b, (short)0, c,
                                                 false, false);
}

// t^(1/(alpha-1)) for t>=0, fast path (v_log_f32 / v_exp_f32)
static __device__ __forceinline__ float pow_inv(float t) {
  return (t > 0.0f) ? __expf(PINV * __logf(t)) : 0.0f;
}

// ---------------------------------------------------------------------------
// Kernel 0: fp32 -> bf16 convert
// ---------------------------------------------------------------------------
__global__ void __launch_bounds__(256) cvt_bf16_kernel(const float* __restrict__ in,
                                                       unsigned short* __restrict__ out,
                                                       int n) {
  int i = blockIdx.x * 256 + threadIdx.x;
  if (i < n) out[i] = f2bf(in[i]);
}

// ---------------------------------------------------------------------------
// QKV / output projections: Y = X * W^T + bias (bf16 WMMA, fp32 accum).
// Block = 128 threads (4 waves). Each wave owns a 16(M) x 64(N) strip with
// 4 accumulators, reusing its A-fragment 4x (2.5 B128 loads per WMMA).
// grid = (M/64, N/64).
//   mode 0: Q  -> [b,h,s,d], scaled by QSCALE
//   mode 1: K  -> [b,h,s,d]
//   mode 2: V  -> [b,h,d,s]  (transposed so P*V B-frags are contiguous)
// ---------------------------------------------------------------------------
static __device__ __forceinline__ void store_qkv_tile(v8f acc, int mb, int mo, int o,
                                                      const float* __restrict__ bias,
                                                      unsigned short* __restrict__ out,
                                                      int mode) {
  const float bv = bias[o];
  const int   h  = o >> 6, d = o & 63;
#pragma unroll
  for (int i = 0; i < 8; ++i) {
    const int m  = mb + i + mo;
    const int bi = m >> 11;            // m / S
    const int s  = m & (SS - 1);
    float v = acc[i] + bv;
    if (mode == 0) {
      v *= QSCALE;
      out[(((size_t)(bi * NHEAD + h)) * SS + s) * DHEAD + d] = f2bf(v);
    } else if (mode == 1) {
      out[(((size_t)(bi * NHEAD + h)) * SS + s) * DHEAD + d] = f2bf(v);
    } else {
      out[(((size_t)(bi * NHEAD + h)) * DHEAD + d) * SS + s] = f2bf(v);
    }
  }
}

__global__ void __launch_bounds__(128) gemm_qkv(const unsigned short* __restrict__ X,
                                                const unsigned short* __restrict__ W,
                                                const float* __restrict__ bias,
                                                unsigned short* __restrict__ out,
                                                int mode) {
  const int lane  = threadIdx.x & 31;
  const int wave  = threadIdx.x >> 5;
  const int mb    = blockIdx.x * 64 + wave * 16;
  const int nb    = blockIdx.y * 64;
  const int c15   = lane & 15;
  const int half8 = (lane >> 4) << 3;

  const unsigned short* xrow = X + (size_t)(mb + c15) * HIDN + half8;
  const unsigned short* wrow = W + (size_t)(nb + c15) * HIDN + half8;

  v8f a0 = {}, a1 = {}, a2 = {}, a3 = {};
#pragma unroll 2
  for (int k0 = 0; k0 < HIDN; k0 += 32) {
    v16bf a  = load_frag(xrow + k0);
    v16bf b0 = load_frag(wrow + k0);
    v16bf b1 = load_frag(wrow + 16 * HIDN + k0);
    v16bf b2 = load_frag(wrow + 32 * HIDN + k0);
    v16bf b3 = load_frag(wrow + 48 * HIDN + k0);
    a0 = wmma_bf16(a, b0, a0);
    a1 = wmma_bf16(a, b1, a1);
    a2 = wmma_bf16(a, b2, a2);
    a3 = wmma_bf16(a, b3, a3);
  }

  const int mo = (lane >> 4) << 3;
  store_qkv_tile(a0, mb, mo, nb +  0 + c15, bias, out, mode);
  store_qkv_tile(a1, mb, mo, nb + 16 + c15, bias, out, mode);
  store_qkv_tile(a2, mb, mo, nb + 32 + c15, bias, out, mode);
  store_qkv_tile(a3, mb, mo, nb + 48 + c15, bias, out, mode);
}

// ---------------------------------------------------------------------------
// Kernel 2: fused scores + entmax-bisect + P*V for one (b,h, 16-query tile).
// grid = (S/16, B*NH), block = 128 (4 waves). Dynamic LDS (~193 KB, CDNA5
// 320KB/WGP makes the whole 16x2048 score tile LDS-resident):
//   xa[16][XA_STRIDE] fp32  (Xa = scores*(alpha-1), via pre-scaled Q)
//   pb[16][PB_STRIDE] bf16  (normalized entmax probabilities)
// ---------------------------------------------------------------------------
__global__ void __launch_bounds__(128) attn_entmax(const unsigned short* __restrict__ Q,
                                                   const unsigned short* __restrict__ K,
                                                   const unsigned short* __restrict__ VT,
                                                   unsigned short* __restrict__ CTX) {
  extern __shared__ char smem[];
  float*          xa = (float*)smem;                                   // 16*XA_STRIDE fp32
  unsigned short* pb = (unsigned short*)(smem + (size_t)16 * XA_STRIDE * 4);

  const int lane  = threadIdx.x & 31;
  const int wave  = threadIdx.x >> 5;
  const int bh    = blockIdx.y;
  const int q0    = blockIdx.x * 16;
  const int c15   = lane & 15;
  const int half8 = (lane >> 4) << 3;
  const int mo    = (lane >> 4) << 3;

  // --- Q fragments for this query tile (K-dim = DHEAD = 2 x 32) ---
  const unsigned short* qrow = Q + ((size_t)bh * SS + q0 + c15) * DHEAD + half8;
  const v16bf aq0 = load_frag(qrow);
  const v16bf aq1 = load_frag(qrow + 32);

  // --- Phase 1: Xa tile = (Q*QSCALE) * K^T, straight into LDS ---
  const unsigned short* kbase = K + (size_t)bh * SS * DHEAD;
#pragma unroll 2
  for (int kt = wave; kt < SS / 16; kt += 4) {
    const int n0 = kt * 16;
    const unsigned short* krow = kbase + (size_t)(n0 + c15) * DHEAD + half8;
    if (n0 + 64 < SS)
      __builtin_prefetch(kbase + (size_t)(n0 + 64 + c15) * DHEAD, 0, 0);
    v16bf bk0 = load_frag(krow);
    v16bf bk1 = load_frag(krow + 32);
    v8f c = {};
    c = wmma_bf16(aq0, bk0, c);
    c = wmma_bf16(aq1, bk1, c);
#pragma unroll
    for (int i = 0; i < 8; ++i)
      xa[(size_t)(i + mo) * XA_STRIDE + n0 + c15] = c[i];
  }
  __syncthreads();

  // --- Phase 2: entmax bisection, 8 lanes per row, 16 rows in parallel.
  //     float2 LDS reads halve ds-op count in the 50-iteration loop. ---
  const int row = threadIdx.x >> 3;
  const int sub = threadIdx.x & 7;
  float*        Xr  = xa + (size_t)row * XA_STRIDE;
  const float2* Xr2 = (const float2*)Xr;

  float mx = -3.4e38f;
  for (int j = sub; j < SS / 2; j += 8) {
    float2 t2 = Xr2[j];
    mx = fmaxf(mx, fmaxf(t2.x, t2.y));
  }
  mx = fmaxf(mx, __shfl_xor(mx, 1));
  mx = fmaxf(mx, __shfl_xor(mx, 2));
  mx = fmaxf(mx, __shfl_xor(mx, 4));

  float tau    = mx - 1.0f;                                   // tau_lo
  float tau_hi = mx - __expf(-ALPHA_M1 * __logf((float)SS));  // mx - (1/d)^(a-1)

  float s0 = 0.0f;
  for (int j = sub; j < SS / 2; j += 8) {
    float2 t2 = Xr2[j];
    s0 += pow_inv(t2.x - tau) + pow_inv(t2.y - tau);
  }
  s0 += __shfl_xor(s0, 1); s0 += __shfl_xor(s0, 2); s0 += __shfl_xor(s0, 4);
  const float f_lo = s0 - 1.0f;

  float dm = tau_hi - tau;
  for (int it = 0; it < NITER; ++it) {
    dm *= 0.5f;
    const float tm = tau + dm;
    float s = 0.0f;
    for (int j = sub; j < SS / 2; j += 8) {
      float2 t2 = Xr2[j];
      s += pow_inv(t2.x - tm) + pow_inv(t2.y - tm);
    }
    s += __shfl_xor(s, 1); s += __shfl_xor(s, 2); s += __shfl_xor(s, 4);
    if ((s - 1.0f) * f_lo >= 0.0f) tau = tm;
  }

  // --- Phase 3: final p = clip(Xa - tau)^inv. Overwrite Xa in place with
  //     unnormalized p during the sum pass (saves an exp/log pair per elem),
  //     then scale + pack to bf16 probabilities. ---
  {
    float2* Xw2 = (float2*)Xr;
    float s = 0.0f;
    for (int j = sub; j < SS / 2; j += 8) {
      float2 t2 = Xr2[j];
      float2 p2;
      p2.x = pow_inv(t2.x - tau);
      p2.y = pow_inv(t2.y - tau);
      s += p2.x + p2.y;
      Xw2[j] = p2;                      // each (row, j) owned by this thread
    }
    s += __shfl_xor(s, 1); s += __shfl_xor(s, 2); s += __shfl_xor(s, 4);
    const float rn = 1.0f / s;
    unsigned int* Pr = (unsigned int*)(pb + (size_t)row * PB_STRIDE);
    for (int j = sub; j < SS / 2; j += 8) {
      float2 p2 = Xr2[j];
      Pr[j] = f2bf_pack2(p2.x * rn, p2.y * rn);
    }
  }
  __syncthreads();

  // --- Phase 4: ctx = P * V, each wave owns a 16-wide d-slice of DH=64 ---
  const int d0 = wave * 16;
  const unsigned short* vrow = VT + ((size_t)bh * DHEAD + d0 + c15) * SS + half8;
  const unsigned short* prow = pb + (size_t)c15 * PB_STRIDE + half8;
  v8f acc = {};
#pragma unroll 4
  for (int k0 = 0; k0 < SS; k0 += 32) {
    v16bf a = load_frag(prow + k0);   // A-frag from LDS probabilities
    v16bf b = load_frag(vrow + k0);   // B-frag from transposed V (contiguous)
    acc = wmma_bf16(a, b, acc);
  }

  const int bq = bh >> 3, hq = bh & 7;
#pragma unroll
  for (int i = 0; i < 8; ++i) {
    const int s_ = q0 + i + mo;
    CTX[((size_t)bq * SS + s_) * HIDN + hq * DHEAD + d0 + c15] = f2bf(acc[i]);
  }
}

// ---------------------------------------------------------------------------
// Kernel 3: out = ctx * Wo^T + bo (fp32 output), same 16x64-per-wave blocking
// ---------------------------------------------------------------------------
__global__ void __launch_bounds__(128) gemm_out(const unsigned short* __restrict__ X,
                                                const unsigned short* __restrict__ W,
                                                const float* __restrict__ bias,
                                                float* __restrict__ out) {
  const int lane  = threadIdx.x & 31;
  const int wave  = threadIdx.x >> 5;
  const int mb    = blockIdx.x * 64 + wave * 16;
  const int nb    = blockIdx.y * 64;
  const int c15   = lane & 15;
  const int half8 = (lane >> 4) << 3;

  const unsigned short* xrow = X + (size_t)(mb + c15) * HIDN + half8;
  const unsigned short* wrow = W + (size_t)(nb + c15) * HIDN + half8;

  v8f a0 = {}, a1 = {}, a2 = {}, a3 = {};
#pragma unroll 2
  for (int k0 = 0; k0 < HIDN; k0 += 32) {
    v16bf a  = load_frag(xrow + k0);
    v16bf b0 = load_frag(wrow + k0);
    v16bf b1 = load_frag(wrow + 16 * HIDN + k0);
    v16bf b2 = load_frag(wrow + 32 * HIDN + k0);
    v16bf b3 = load_frag(wrow + 48 * HIDN + k0);
    a0 = wmma_bf16(a, b0, a0);
    a1 = wmma_bf16(a, b1, a1);
    a2 = wmma_bf16(a, b2, a2);
    a3 = wmma_bf16(a, b3, a3);
  }

  const int mo = (lane >> 4) << 3;
#pragma unroll
  for (int i = 0; i < 8; ++i) {
    const int m = mb + i + mo;
    float* orow = out + (size_t)m * HIDN;
    orow[nb +  0 + c15] = a0[i] + bias[nb +  0 + c15];
    orow[nb + 16 + c15] = a1[i] + bias[nb + 16 + c15];
    orow[nb + 32 + c15] = a2[i] + bias[nb + 32 + c15];
    orow[nb + 48 + c15] = a3[i] + bias[nb + 48 + c15];
  }
}

// ---------------------------------------------------------------------------
// Host-side launcher
// ---------------------------------------------------------------------------
extern "C" void kernel_launch(void* const* d_in, const int* in_sizes, int n_in,
                              void* d_out, int out_size, void* d_ws, size_t ws_size,
                              hipStream_t stream) {
  (void)in_sizes; (void)n_in; (void)out_size; (void)ws_size;

  const float* hidden = (const float*)d_in[0];
  const float* Wq = (const float*)d_in[1];
  const float* bq = (const float*)d_in[2];
  const float* Wk = (const float*)d_in[3];
  const float* bk = (const float*)d_in[4];
  const float* Wv = (const float*)d_in[5];
  const float* bv = (const float*)d_in[6];
  const float* Wo = (const float*)d_in[7];
  const float* bo = (const float*)d_in[8];
  float* out = (float*)d_out;

  char*  ws  = (char*)d_ws;
  size_t off = 0;
  auto take = [&](size_t bytes) -> char* {
    char* p = ws + off;
    off += (bytes + 255) & ~(size_t)255;
    return p;
  };

  const size_t nX   = (size_t)BB * SS * HIDN;      // 4,194,304
  const size_t nW   = (size_t)HIDN * HIDN;         // 262,144
  const size_t nQKV = (size_t)BB * NHEAD * SS * DHEAD;

  unsigned short* xbf   = (unsigned short*)take(nX * 2);
  unsigned short* wqb   = (unsigned short*)take(nW * 2);
  unsigned short* wkb   = (unsigned short*)take(nW * 2);
  unsigned short* wvb   = (unsigned short*)take(nW * 2);
  unsigned short* wob   = (unsigned short*)take(nW * 2);
  unsigned short* qws   = (unsigned short*)take(nQKV * 2);
  unsigned short* kws   = (unsigned short*)take(nQKV * 2);
  unsigned short* vtws  = (unsigned short*)take(nQKV * 2);
  unsigned short* ctxws = (unsigned short*)take(nX * 2);

  cvt_bf16_kernel<<<(int)((nX + 255) / 256), 256, 0, stream>>>(hidden, xbf, (int)nX);
  cvt_bf16_kernel<<<(int)((nW + 255) / 256), 256, 0, stream>>>(Wq, wqb, (int)nW);
  cvt_bf16_kernel<<<(int)((nW + 255) / 256), 256, 0, stream>>>(Wk, wkb, (int)nW);
  cvt_bf16_kernel<<<(int)((nW + 255) / 256), 256, 0, stream>>>(Wv, wvb, (int)nW);
  cvt_bf16_kernel<<<(int)((nW + 255) / 256), 256, 0, stream>>>(Wo, wob, (int)nW);

  dim3 gg(128, 8);  // M/64, N/64
  gemm_qkv<<<gg, 128, 0, stream>>>(xbf, wqb, bq, qws, 0);
  gemm_qkv<<<gg, 128, 0, stream>>>(xbf, wkb, bk, kws, 1);
  gemm_qkv<<<gg, 128, 0, stream>>>(xbf, wvb, bv, vtws, 2);

  const size_t smem = (size_t)16 * XA_STRIDE * 4 + (size_t)16 * PB_STRIDE * 2;
  (void)hipFuncSetAttribute((const void*)attn_entmax,
                            hipFuncAttributeMaxDynamicSharedMemorySize, (int)smem);
  attn_entmax<<<dim3(SS / 16, BB * NHEAD), 128, smem, stream>>>(qws, kws, vtws, ctxws);

  gemm_out<<<gg, 128, 0, stream>>>(ctxws, wob, bo, out);
}
